// WindowMSAV2_48911087566867
// MI455X (gfx1250) — matrix operand order repaired and checked by reading, not compiled, not run on hardware
//
#include <hip/hip_runtime.h>
#include <hip/hip_bf16.h>

typedef __bf16 bf16;
typedef __attribute__((ext_vector_type(16))) __bf16 v16bf;
typedef __attribute__((ext_vector_type(8)))  float  v8f;

typedef unsigned int       u32;
typedef unsigned long long u64;
typedef __attribute__((ext_vector_type(4))) u32 u32x4;
typedef __attribute__((ext_vector_type(8))) int i32x8;
typedef __attribute__((ext_vector_type(4))) int i32x4;

// TDM path: ROCm 7.2 (clang-22) 5-arg tensor_load_to_lds, probe-verified arity.
#if defined(__gfx1250__) && defined(__has_builtin)
#if __has_builtin(__builtin_amdgcn_tensor_load_to_lds) && \
    __has_builtin(__builtin_amdgcn_s_wait_tensorcnt) && (__clang_major__ < 23)
#define USE_TDM 1
#endif
#endif
#ifndef USE_TDM
#define USE_TDM 0
#endif

// ---------------------------------------------------------------------------
// WMMA helpers (CDNA5 v_wmma_f32_16x16x32_bf16, wave32)
// A (16x32, MxK): lane l -> m = l&15, khalf = l>>4. VGPR v holds K pair
//   k = 2v + khalf*16   (matches B order)
// B (32x16, KxN): lane l -> n = l&15, lanes 16-31 take K+16. VGPR v holds
//   k = 2v + (l>>4)*16
// C/D (16x16 f32): lane l -> n = l&15, rows m = (l>>4)*8 + v
// ---------------------------------------------------------------------------
__device__ __forceinline__ v16bf load_A16x32(const bf16* __restrict__ src, int ld) {
    const int l = threadIdx.x & 31;
    const int m = l & 15;
    const int khalf = (l >> 4) * 8;
    v16bf a;
#pragma unroll
    for (int v = 0; v < 8; ++v) {
        const int k = 2 * v + khalf * 2;
        a[2 * v]     = src[m * ld + k];
        a[2 * v + 1] = src[m * ld + k + 1];
    }
    return a;
}

__device__ __forceinline__ v16bf load_B32x16(const bf16* __restrict__ src,
                                             int strideK, int strideN) {
    const int l = threadIdx.x & 31;
    const int n = (l & 15) * strideN;
    const int kh = (l >> 4) * 16;
    v16bf b;
#pragma unroll
    for (int v = 0; v < 8; ++v) {
        const int k = kh + 2 * v;
        b[2 * v]     = src[k * strideK + n];
        b[2 * v + 1] = src[(k + 1) * strideK + n];
    }
    return b;
}

__device__ __forceinline__ v8f wmma_bf16(v16bf a, v16bf b, v8f c) {
    return __builtin_amdgcn_wmma_f32_16x16x32_bf16(
        false, a, false, b, (short)0, c, false, false);
}

__device__ __forceinline__ void store_D_f32(float* __restrict__ dst, int ld, v8f d) {
    const int l = threadIdx.x & 31;
    const int n = l & 15;
    const int mb = (l >> 4) * 8;
#pragma unroll
    for (int v = 0; v < 8; ++v) dst[(mb + v) * ld + n] = d[v];
}

__device__ __forceinline__ void store_D_bf16(bf16* __restrict__ dst, int ld, v8f d) {
    const int l = threadIdx.x & 31;
    const int n = l & 15;
    const int mb = (l >> 4) * 8;
#pragma unroll
    for (int v = 0; v < 8; ++v) dst[(mb + v) * ld + n] = (bf16)d[v];
}

// ---------------------------------------------------------------------------
// Kernel A: log-CPB bias -> 16*sigmoid(bias)[4][64][64] f32, + head scales.
// ---------------------------------------------------------------------------
__global__ __launch_bounds__(256) void cpb_bias_kernel(
    const float* __restrict__ cpb_w1, const float* __restrict__ cpb_b1,
    const float* __restrict__ cpb_w2, const float* __restrict__ logit_scale,
    float* __restrict__ bias_out, float* __restrict__ scale_out) {
    __shared__ float table[225 * 4];
    const int t = threadIdx.x;
    if (t < 225) {
        float ch = (float)(t / 15) - 7.0f;
        float cw = (float)(t % 15) - 7.0f;
        float v0 = ch * (8.0f / 7.0f);
        float v1 = cw * (8.0f / 7.0f);
        float s0 = (v0 > 0.f) ? 1.f : ((v0 < 0.f) ? -1.f : 0.f);
        float s1 = (v1 > 0.f) ? 1.f : ((v1 < 0.f) ? -1.f : 0.f);
        float c0 = s0 * log2f(fabsf(v0) + 1.0f) * (1.0f / 3.0f);
        float c1 = s1 * log2f(fabsf(v1) + 1.0f) * (1.0f / 3.0f);
        float acc0 = 0.f, acc1 = 0.f, acc2 = 0.f, acc3 = 0.f;
        for (int i = 0; i < 512; ++i) {
            float h = fmaxf(c0 * cpb_w1[2 * i] + c1 * cpb_w1[2 * i + 1] + cpb_b1[i], 0.f);
            acc0 += h * cpb_w2[0 * 512 + i];
            acc1 += h * cpb_w2[1 * 512 + i];
            acc2 += h * cpb_w2[2 * 512 + i];
            acc3 += h * cpb_w2[3 * 512 + i];
        }
        table[t * 4 + 0] = acc0;
        table[t * 4 + 1] = acc1;
        table[t * 4 + 2] = acc2;
        table[t * 4 + 3] = acc3;
    }
    if (t < 4) scale_out[t] = __expf(fminf(logit_scale[t], logf(100.0f)));
    __syncthreads();
    for (int idx = t; idx < 4 * 64 * 64; idx += 256) {
        const int h = idx >> 12;
        const int n = (idx >> 6) & 63;
        const int m = idx & 63;
        const int d0 = (n >> 3) - (m >> 3) + 7;
        const int d1 = (n & 7) - (m & 7) + 7;
        const float bv = table[(d0 * 15 + d1) * 4 + h];
        bias_out[idx] = 16.0f / (1.0f + __expf(-bv));
    }
}

// ---------------------------------------------------------------------------
// Kernel B: fused window attention. One workgroup (8 wave32) per window.
// LDS (225.5 KB dynamic):
//   [0      , 16384 )  xb    : x tile     64x128 bf16
//   [16384  , 65536 )  lW    : weights    192x128 bf16 (qkv halves / proj)
//   [65536  , 98304 )  lQK   : q|k        64x256 bf16 (norm+scale folded)
//   [98304  , 114688)  lVt   : V^T        4 x 32dim x 64tok bf16
//   [114688 , 180224)  lS    : scores     4x64x64 f32 (phase1: x f32 TDM stage)
//   [180224 , 212992)  lP    : softmax P  4x64x64 bf16
//   [212992 , 229376)  lO    : attn out   64x128 bf16
//   [229376 , 230912)  lBias : fused qkv bias 384 f32
// ---------------------------------------------------------------------------
#define SMEM_BYTES 230912

__global__ __launch_bounds__(256) void swin_attn_kernel(
    const float* __restrict__ x, const float* __restrict__ mask,
    const float* __restrict__ qkv_w, const float* __restrict__ q_bias,
    const float* __restrict__ v_bias,
    const float* __restrict__ proj_w, const float* __restrict__ proj_b,
    const float* __restrict__ bias_tab, const float* __restrict__ head_scale,
    float* __restrict__ out) {
    extern __shared__ char smem[];
    bf16*  xb    = (bf16*) (smem);
    bf16*  lW    = (bf16*) (smem + 16384);
    bf16*  lQK   = (bf16*) (smem + 65536);
    bf16*  lVt   = (bf16*) (smem + 98304);
    float* lS    = (float*)(smem + 114688);
    bf16*  lP    = (bf16*) (smem + 180224);
    bf16*  lO    = (bf16*) (smem + 212992);
    float* lBias = (float*)(smem + 229376);

    const int tid  = threadIdx.x;
    const int lane = tid & 31;
    const int wave = tid >> 5;
    const int blk  = blockIdx.x;

    const float* xg = x + (size_t)blk * 64 * 128;

    // ---- Phase 1: stage x. TDM path: one 32KB 1-D tensor DMA into lS. ----
#if USE_TDM
    if (tid == 0) {
        const u64 ga = (u64)(uintptr_t)xg;
        const u32 ldsoff = (u32)(size_t)(smem + 114688);
        u32x4 g0;
        g0[0] = 1u;                                   // count=1, user mode
        g0[1] = ldsoff;                               // lds_addr
        g0[2] = (u32)ga;                              // global_addr[31:0]
        g0[3] = (u32)((ga >> 32) & 0x1FFFFFFu) | (2u << 30);  // addr[56:32]|type=2
        i32x8 g1;
        const u32 td0 = 8192;                         // tensor_dim0 (f32 elems)
        g1[0] = (int)(2u << 16);                      // data_size = 4B
        g1[1] = (int)((td0 & 0xFFFFu) << 16);         // tensor_dim0[15:0]
        g1[2] = (int)((td0 >> 16) | (1u << 16));      // tensor_dim0[31:16], dim1=1
        g1[3] = (int)(8192u << 16);                   // tile_dim0 = 8192
        g1[4] = 0;                                    // tile_dim1=0, tile_dim2=0
        g1[5] = (int)td0;                             // tensor_dim0_stride
        g1[6] = 0;
        g1[7] = 0;
        i32x4 z4 = {0, 0, 0, 0};
        __builtin_amdgcn_tensor_load_to_lds(g0, g1, z4, z4, 0);
    }
#endif

    // warm proj_w into L2/WGP$ for phase 7
    __builtin_prefetch(proj_w + tid * 64, 0, 1);

    // fused qkv bias -> LDS (kills divergent loads in the QKV tile loop)
    for (int i = tid; i < 384; i += 256)
        lBias[i] = (i < 128) ? q_bias[i] : ((i >= 256) ? v_bias[i - 256] : 0.0f);

    // stage qkv_w half 0 while the DMA flies
    for (int i = tid; i < 192 * 128; i += 256) lW[i] = (bf16)qkv_w[i];

#if USE_TDM
    if (tid == 0) __builtin_amdgcn_s_wait_tensorcnt((unsigned short)0);
    __syncthreads();
    for (int i = tid; i < 64 * 128; i += 256) xb[i] = (bf16)lS[i];
#else
    for (int i = tid; i < 64 * 128; i += 256) xb[i] = (bf16)xg[i];
#endif
    __syncthreads();

    // ---- Phase 2: QKV = x @ qkv_w^T + bias (two 192-col halves) ----
    for (int half = 0; half < 2; ++half) {
        if (half == 1) {
            const float* wg = qkv_w + (size_t)192 * 128;
            for (int i = tid; i < 192 * 128; i += 256) lW[i] = (bf16)wg[i];
            __syncthreads();
        }
        for (int t = wave; t < 48; t += 8) {          // 4 M-tiles x 12 N-tiles
            const int mt = t / 12, nt = t % 12;
            v8f acc = {};
#pragma unroll
            for (int kt = 0; kt < 4; ++kt) {
                v16bf a  = load_A16x32(xb + mt * 16 * 128 + kt * 32, 128);
                v16bf bm = load_B32x16(lW + nt * 16 * 128 + kt * 32, 1, 128);
                acc = wmma_bf16(a, bm, acc);
            }
            const int j0 = half * 192 + nt * 16;
            const float bv = lBias[j0 + (lane & 15)];
#pragma unroll
            for (int v = 0; v < 8; ++v) acc[v] += bv;
            if (j0 < 256) {
                // q | k: token-major
                store_D_bf16(lQK + mt * 16 * 256 + j0, 256, acc);
            } else {
                // V: transpose-store -> lVt[head][dim][token] (wave-uniform branch)
                const int jv = j0 - 256;
                const int h = jv >> 5;
                const int dbase = jv & 31;
                const int n = lane & 15;
                const int mb = (lane >> 4) * 8;
#pragma unroll
                for (int v = 0; v < 8; ++v)
                    lVt[h * 2048 + (dbase + n) * 64 + mt * 16 + mb + v] = (bf16)acc[v];
            }
        }
        __syncthreads();
    }

    // ---- Phase 3: fold cosine norm (+logit scale into q) per row ----
    for (int r = tid; r < 512; r += 256) {
        const int isK = r >> 8;
        const int h = (r >> 6) & 3;
        const int n = r & 63;
        bf16* p = lQK + n * 256 + isK * 128 + h * 32;
        float vals[32];
        float ss = 0.f;
#pragma unroll
        for (int d = 0; d < 32; ++d) { vals[d] = (float)p[d]; ss += vals[d] * vals[d]; }
        float inv = 1.0f / fmaxf(sqrtf(ss), 1e-12f);
        if (!isK) inv *= head_scale[h];
#pragma unroll
        for (int d = 0; d < 32; ++d) p[d] = (bf16)(vals[d] * inv);
    }
    __syncthreads();

    // ---- Phase 4: S_h = q'_h @ k'_h^T (per head 64x64, K=32) ----
    for (int t = wave; t < 64; t += 8) {
        const int h = t >> 4, mt = (t >> 2) & 3, nt = t & 3;
        v16bf a  = load_A16x32(lQK + (mt * 16) * 256 + h * 32, 256);
        v16bf bm = load_B32x16(lQK + (nt * 16) * 256 + 128 + h * 32, 1, 256);
        v8f acc = {};
        acc = wmma_bf16(a, bm, acc);
        store_D_f32(lS + h * 4096 + mt * 16 * 64 + nt * 16, 64, acc);
    }
    __syncthreads();

    // ---- Phase 5: softmax rows (+CPB bias +shift mask); 256 rows exactly ----
    {
        const int h = tid >> 6, row = tid & 63;
        const float* brow = bias_tab + h * 4096 + row * 64;
        const float* mrow = mask + (size_t)(blk & 255) * 4096 + row * 64;
        const float* srow = lS + h * 4096 + row * 64;
        float vals[64];
        float mx = -1e30f;
#pragma unroll
        for (int c = 0; c < 64; ++c) {
            float v = srow[c] + brow[c] + mrow[c];
            vals[c] = v;
            mx = fmaxf(mx, v);
        }
        float sum = 0.f;
#pragma unroll
        for (int c = 0; c < 64; ++c) {
            float e = __expf(vals[c] - mx);
            vals[c] = e;
            sum += e;
        }
        const float inv = 1.0f / sum;
        bf16* prow = lP + h * 4096 + row * 64;
#pragma unroll
        for (int c = 0; c < 64; ++c) prow[c] = (bf16)(vals[c] * inv);
    }
    __syncthreads();

    // ---- Phase 6: O_h = P_h @ V_h via V^T (strideK=1 -> b128 fragments) ----
    for (int t = wave; t < 32; t += 8) {
        const int h = t >> 3, mt = (t >> 1) & 3, nt = t & 1;
        v8f acc = {};
#pragma unroll
        for (int kt = 0; kt < 2; ++kt) {
            v16bf a  = load_A16x32(lP + h * 4096 + mt * 16 * 64 + kt * 32, 64);
            v16bf bm = load_B32x16(lVt + h * 2048 + nt * 16 * 64 + kt * 32, 1, 64);
            acc = wmma_bf16(a, bm, acc);
        }
        store_D_bf16(lO + mt * 16 * 128 + h * 32 + nt * 16, 128, acc);
    }
    __syncthreads();

    // ---- Phase 7: out = O @ proj_w^T + proj_b ----
    for (int i = tid; i < 128 * 128; i += 256) lW[i] = (bf16)proj_w[i];
    __syncthreads();
    float* og = out + (size_t)blk * 64 * 128;
    for (int t = wave; t < 32; t += 8) {
        const int mt = t >> 3, nt = t & 7;
        v8f acc = {};
#pragma unroll
        for (int kt = 0; kt < 4; ++kt) {
            v16bf a  = load_A16x32(lO + mt * 16 * 128 + kt * 32, 128);
            v16bf bm = load_B32x16(lW + nt * 16 * 128 + kt * 32, 1, 128);
            acc = wmma_bf16(a, bm, acc);
        }
        const int n = lane & 15;
        const int mb = (lane >> 4) * 8;
        const float pb = proj_b[nt * 16 + n];
#pragma unroll
        for (int v = 0; v < 8; ++v)
            og[(mt * 16 + mb + v) * 128 + nt * 16 + n] = acc[v] + pb;
    }
}

extern "C" void kernel_launch(void* const* d_in, const int* in_sizes, int n_in,
                              void* d_out, int out_size, void* d_ws, size_t ws_size,
                              hipStream_t stream) {
    const float* x           = (const float*)d_in[0];
    const float* mask        = (const float*)d_in[1];
    const float* qkv_w       = (const float*)d_in[2];
    const float* q_bias      = (const float*)d_in[3];
    const float* v_bias      = (const float*)d_in[4];
    const float* logit_scale = (const float*)d_in[5];
    const float* cpb_w1      = (const float*)d_in[6];
    const float* cpb_b1      = (const float*)d_in[7];
    const float* cpb_w2      = (const float*)d_in[8];
    const float* proj_w      = (const float*)d_in[9];
    const float* proj_b      = (const float*)d_in[10];
    float* out = (float*)d_out;

    float* bias_tab   = (float*)d_ws;            // 4*64*64 f32
    float* head_scale = bias_tab + 4 * 64 * 64;  // 4 f32

    cpb_bias_kernel<<<1, 256, 0, stream>>>(cpb_w1, cpb_b1, cpb_w2, logit_scale,
                                           bias_tab, head_scale);

    static bool attr_set = false;
    if (!attr_set) {
        hipFuncSetAttribute(reinterpret_cast<const void*>(swin_attn_kernel),
                            hipFuncAttributeMaxDynamicSharedMemorySize, SMEM_BYTES);
        attr_set = true;
    }
    swin_attn_kernel<<<8192, 256, SMEM_BYTES, stream>>>(
        x, mask, qkv_w, q_bias, v_bias, proj_w, proj_b, bias_tab, head_scale, out);
}
